// QuantizedLinear_64630667870406
// MI455X (gfx1250) — 8-bit matrix operand order repaired and checked by reading, compiled, not run on hardware
//
#include <hip/hip_runtime.h>
#include <stdint.h>

#define TOKENS 8192
#define IN_F   4096
#define OUT_F  4096

typedef __attribute__((ext_vector_type(8))) int v8i;

__global__ __launch_bounds__(256) void pack_i32_to_i8(const int* __restrict__ src,
                                                      uint32_t* __restrict__ dst,
                                                      int n4) {
  int i = blockIdx.x * blockDim.x + threadIdx.x;
  if (i >= n4) return;
  int4 v = ((const int4*)src)[i];
  uint32_t p = (uint32_t)(v.x & 0xff) |
               ((uint32_t)(v.y & 0xff) << 8) |
               ((uint32_t)(v.z & 0xff) << 16) |
               ((uint32_t)(v.w & 0xff) << 24);
  dst[i] = p;
}

__global__ __launch_bounds__(256) void qlinear_wmma_iu8(
    const uint8_t* __restrict__ Ap,
    const uint8_t* __restrict__ Wp,
    const int*     __restrict__ bias,
    const float*   __restrict__ wscale,
    const float*   __restrict__ s_in,
    const float*   __restrict__ s_out,
    const float*   __restrict__ zp,
    float*         __restrict__ out) {
  __shared__ __align__(16) uint8_t lds_a[128][64];
  __shared__ __align__(16) uint8_t lds_b[128][64];

  const int tid    = threadIdx.x;
  const int lane   = tid & 31;
  const int wid    = tid >> 5;
  const int wave_m = wid & 3;
  const int wave_n = wid >> 2;

  const int block_n = blockIdx.x * 128;
  const int block_m = blockIdx.y * 128;

  const int row = tid >> 1;
  const int seg = (tid & 1) * 32;

  const int half = lane >> 4;
  const int l16  = lane & 15;

  v8i acc[2][4];
  const v8i vzero = {0, 0, 0, 0, 0, 0, 0, 0};
#pragma unroll
  for (int mt = 0; mt < 2; ++mt)
#pragma unroll
    for (int nt = 0; nt < 4; ++nt)
      acc[mt][nt] = vzero;

  for (int kb = 0; kb < IN_F; kb += 64) {
    const uint8_t* ga = Ap + (size_t)(block_m + row) * IN_F + kb + seg;
    const uint8_t* gb = Wp + (size_t)(block_n + row) * IN_F + kb + seg;
    const int4 av0 = ((const int4*)ga)[0];
    const int4 av1 = ((const int4*)ga)[1];
    const int4 bv0 = ((const int4*)gb)[0];
    const int4 bv1 = ((const int4*)gb)[1];

    __syncthreads();
    *(int4*)&lds_a[row][seg]      = av0;
    *(int4*)&lds_a[row][seg + 16] = av1;
    *(int4*)&lds_b[row][seg]      = bv0;
    *(int4*)&lds_b[row][seg + 16] = bv1;
    __syncthreads();

    v8i afr[2];
    const int k0a = half * 16;
#pragma unroll
    for (int mt = 0; mt < 2; ++mt) {
      const uint8_t* ar = &lds_a[wave_m * 32 + mt * 16 + l16][0];
      const int2 p0 = *(const int2*)(ar + k0a);
      const int2 p1 = *(const int2*)(ar + k0a + 8);
      const int2 p2 = *(const int2*)(ar + k0a + 32);
      const int2 p3 = *(const int2*)(ar + k0a + 40);
      v8i a;
      a[0] = p0.x; a[1] = p0.y;
      a[2] = p1.x; a[3] = p1.y;
      a[4] = p2.x; a[5] = p2.y;
      a[6] = p3.x; a[7] = p3.y;
      afr[mt] = a;
    }

    const int kb0 = half * 16;
#pragma unroll
    for (int nt = 0; nt < 4; ++nt) {
      const uint8_t* br = &lds_b[wave_n * 64 + nt * 16 + l16][0];
      const int4 q0 = *(const int4*)(br + kb0);
      const int4 q1 = *(const int4*)(br + kb0 + 32);
      v8i b;
      b[0] = q0.x; b[1] = q0.y; b[2] = q0.z; b[3] = q0.w;
      b[4] = q1.x; b[5] = q1.y; b[6] = q1.z; b[7] = q1.w;
#pragma unroll
      for (int mt = 0; mt < 2; ++mt) {
        acc[mt][nt] = __builtin_amdgcn_wmma_i32_16x16x64_iu8(
            true, afr[mt], true, b, acc[mt][nt],
             false,  false);
      }
    }
  }

  const float smul = (*s_in) / (*s_out);
  const float ozp  = *zp;
#pragma unroll
  for (int nt = 0; nt < 4; ++nt) {
    const int n    = block_n + wave_n * 64 + nt * 16 + l16;
    const float sc = wscale[n] * smul;
    const float bz = (float)bias[n];
#pragma unroll
    for (int mt = 0; mt < 2; ++mt) {
      const int mbase = block_m + wave_m * 32 + mt * 16 + half * 8;
#pragma unroll
      for (int r = 0; r < 8; ++r) {
        float y = sc * ((float)acc[mt][nt][r] + bz) + ozp;
        y = __builtin_rintf(y);
        y = fminf(127.0f, fmaxf(-128.0f, y));
        out[(size_t)(mbase + r) * OUT_F + n] = y;
      }
    }
  }
}

extern "C" void kernel_launch(void* const* d_in, const int* in_sizes, int n_in,
                              void* d_out, int out_size, void* d_ws, size_t ws_size,
                              hipStream_t stream) {
  const int*   q_x    = (const int*)d_in[0];
  const int*   q_w    = (const int*)d_in[1];
  const int*   bias   = (const int*)d_in[2];
  const float* wscale = (const float*)d_in[3];
  const float* s_in   = (const float*)d_in[4];
  const float* s_out  = (const float*)d_in[5];
  const float* zp     = (const float*)d_in[6];
  float*       out    = (float*)d_out;

  uint8_t* Ap = (uint8_t*)d_ws;
  uint8_t* Wp = Ap + (size_t)TOKENS * IN_F;

  const int n4_a = TOKENS * IN_F / 4;
  const int n4_w = OUT_F * IN_F / 4;
  pack_i32_to_i8<<<(n4_a + 255) / 256, 256, 0, stream>>>(q_x, (uint32_t*)Ap, n4_a);
  pack_i32_to_i8<<<(n4_w + 255) / 256, 256, 0, stream>>>(q_w, (uint32_t*)Wp, n4_w);

  dim3 grid(OUT_F / 128, TOKENS / 128);
  qlinear_wmma_iu8<<<grid, 256, 0, stream>>>(Ap, Wp, bias, wscale, s_in, s_out, zp, out);
}
